// LinearAttention_5377299054788
// MI455X (gfx1250) — compile-verified, matrix operand order checked by reading, not run on hardware
//
#include <hip/hip_runtime.h>
#include <hip/hip_bf16.h>
#include <cstdint>

// ---------------------------------------------------------------------------
// LinearAttention for MI455X (gfx1250):
//   prep:   x -> bf16, W -> transposed bf16 (TDM/WMMA-friendly layouts)
//   GEMMs:  TDM (tensor_load_to_lds) double-buffered LDS tiles feeding
//           v_wmma_f32_16x16x32_bf16, fused RoPE/elu/bias epilogues
//   core:   fp32 LDS-tiled kv-reduction + context/norm kernels
// ---------------------------------------------------------------------------

typedef __attribute__((ext_vector_type(16))) __bf16        v16bf;
typedef __attribute__((ext_vector_type(8)))  float         v8f;
typedef __attribute__((ext_vector_type(4)))  unsigned int  u32x4;
typedef __attribute__((ext_vector_type(8)))  int           i32x8;
typedef __attribute__((ext_vector_type(4)))  int           i32x4;

#define NHEADS 16
#define SEQ_L  8192
#define DHEAD  64
#define EMB    1024
#define EPS_LA 1e-4f

#if defined(__has_builtin)
#if __has_builtin(__builtin_amdgcn_tensor_load_to_lds)
#define HAVE_TDM 1
#endif
#endif
#ifndef HAVE_TDM
#define HAVE_TDM 0
#endif

__device__ __forceinline__ unsigned short f32_to_bf16_bits(float f) {
  union { float f; unsigned u; } v; v.f = f;
  unsigned r = v.u + 0x7FFFu + ((v.u >> 16) & 1u);   // round-to-nearest-even
  return (unsigned short)(r >> 16);
}

#if HAVE_TDM
// ---------------------------------------------------------------------------
// Post one 2D TDM descriptor: copy a 128(row) x 32(col) bf16 tile from a
// row-major tensor (row pitch = stride0 elements) into LDS, padding each
// 64B row with +32B so the LDS pitch is 96B (48 shorts) for bank spreading.
// D# packing per CDNA5 ISA 8.3/8.4 (group0: count/lds/global/type,
// group1: data_size|pad | dims | tile dims | stride). Groups 2/3 are zero
// for a 2D tile; the toolchain's 6-arg builtin takes an extra (zeroed)
// int32x8 operand before the cache-policy immediate.
// ---------------------------------------------------------------------------
__device__ __forceinline__ void tdm_load_tile_2d(unsigned lds_addr,
                                                 unsigned long long gaddr,
                                                 unsigned tensor_d0,
                                                 unsigned tensor_d1,
                                                 unsigned stride0)
{
  u32x4 g0;
  g0[0] = 1u;                                            // count=1, user D#
  g0[1] = lds_addr;                                      // lds_addr (bytes)
  g0[2] = (unsigned)gaddr;                               // global_addr[31:0]
  g0[3] = (unsigned)((gaddr >> 32) & 0x01FFFFFFull)      // global_addr[56:32]
          | (2u << 30);                                  // type = 2 ("image")
  i32x8 g1;
  g1[0] = (int)((1u << 16)                               // data_size = 2B
              | (1u << 20)                               // pad_enable
              | (3u << 22)                               // pad_interval: 64B
              | (7u << 25));                             // pad_amount : 32B
  g1[1] = (int)((tensor_d0 & 0xFFFFu) << 16);            // tensor_dim0 lo
  g1[2] = (int)((tensor_d0 >> 16) | ((tensor_d1 & 0xFFFFu) << 16));
  g1[3] = (int)((tensor_d1 >> 16) | (32u << 16));        // tile_dim0 = 32
  g1[4] = (int)128u;                                     // tile_dim1=128, d2=0
  g1[5] = (int)stride0;                                  // dim0 stride lo
  g1[6] = 0;                                             // stride hi / d1strd
  g1[7] = 0;
  const i32x4 gz4 = {0, 0, 0, 0};
  const i32x8 gz8 = {0, 0, 0, 0, 0, 0, 0, 0};
  __builtin_amdgcn_tensor_load_to_lds(g0, g1, gz4, gz4, gz8, 0);
}
#endif

// ---------------------------------------------------------------------------
// Prep kernels: fp32 -> bf16 elementwise; fp32 [K][N] -> bf16 [N][K] transpose
// ---------------------------------------------------------------------------
__global__ __launch_bounds__(256)
void cvt_bf16_kernel(const float* __restrict__ in, unsigned short* __restrict__ out,
                     long long n4)
{
  const long long i = (long long)blockIdx.x * 256 + threadIdx.x;
  if (i < n4) {
    const float4 v = ((const float4*)in)[i];
    ushort4 o;
    o.x = f32_to_bf16_bits(v.x); o.y = f32_to_bf16_bits(v.y);
    o.z = f32_to_bf16_bits(v.z); o.w = f32_to_bf16_bits(v.w);
    ((ushort4*)out)[i] = o;
  }
}

__global__ __launch_bounds__(256)
void transpose_cvt_kernel(const float* __restrict__ in, unsigned short* __restrict__ out,
                          int K, int N)   // in: [K][N] fp32 -> out: [N][K] bf16
{
  __shared__ unsigned short t[32][33];
  const int nb = blockIdx.x * 32;
  const int kb = blockIdx.y * 32;
  const int tx = threadIdx.x & 31;
  const int ty = threadIdx.x >> 5;                 // 8 rows per pass
#pragma unroll
  for (int i = 0; i < 32; i += 8)
    t[ty + i][tx] = f32_to_bf16_bits(in[(size_t)(kb + ty + i) * N + nb + tx]);
  __syncthreads();
#pragma unroll
  for (int i = 0; i < 32; i += 8)
    out[(size_t)(nb + ty + i) * K + kb + tx] = t[tx][ty + i];
}

// ---------------------------------------------------------------------------
// bf16 WMMA GEMM: C[M,N] = A[M,K] * Bt[N,K]^T (+bias), fused epilogue.
//  Block tile 128x128, K-step 32, 8 waves in 2(M)x4(N); each wave 64x32 =
//  4x2 v_wmma_f32_16x16x32_bf16 frags. LDS tiles double-buffered; filled by
//  the Tensor Data Mover (wave 0 posts descriptors, s_wait_tensorcnt +
//  workgroup barrier publish them) -- VALU stays free for WMMA + epilogue.
// EPILOGUE 0: bias + RoPE (xor-shuffle pairing) + elu+1 + mask -> q'/k'/v
// EPILOGUE 1: bias -> fp32 outp
// ---------------------------------------------------------------------------
template <int EPILOGUE>
__global__ __launch_bounds__(256, 2)
void wmma_gemm_tdm_kernel(const unsigned short* __restrict__ Ag,   // [M][K] bf16
                          const unsigned short* __restrict__ Btg,  // [N][K] bf16
                          const float* __restrict__ bias,
                          const float* __restrict__ fcos, const float* __restrict__ fsin,
                          const float* __restrict__ mask,
                          float* __restrict__ qp, float* __restrict__ kp,
                          float* __restrict__ vp, float* __restrict__ outp,
                          int M, int N, int K)
{
  __shared__ __align__(32) unsigned short Asl[2][128 * 48];
  __shared__ __align__(32) unsigned short Bsl[2][128 * 48];

  const int tid  = threadIdx.x;
  const int lane = tid & 31;
  const int wm   = (tid >> 5) >> 2;            // 0..1 : 64-row slice
  const int wn   = (tid >> 5) & 3;             // 0..3 : 32-col slice
  const int mBase = blockIdx.y * 128;
  const int nBase = blockIdx.x * 128;
  const int wave_id = __builtin_amdgcn_readfirstlane(tid >> 5);  // uniform

  const v8f zero = {0.f, 0.f, 0.f, 0.f, 0.f, 0.f, 0.f, 0.f};
  v8f c[4][2];
#pragma unroll
  for (int i = 0; i < 4; ++i)
#pragma unroll
    for (int j = 0; j < 2; ++j) c[i][j] = zero;

  const int KT = K >> 5;

#if HAVE_TDM
  const unsigned long long aBase =
      (unsigned long long)(uintptr_t)Ag + (unsigned long long)mBase * K * 2ull;
  const unsigned long long bBase =
      (unsigned long long)(uintptr_t)Btg + (unsigned long long)nBase * K * 2ull;
  // generic LDS pointers: low 32 bits are the wave-relative LDS byte address
  const unsigned ldsA[2] = {(unsigned)(uintptr_t)&Asl[0][0],
                            (unsigned)(uintptr_t)&Asl[1][0]};
  const unsigned ldsB[2] = {(unsigned)(uintptr_t)&Bsl[0][0],
                            (unsigned)(uintptr_t)&Bsl[1][0]};
  if (wave_id == 0) {                          // scalar branch: one wave issues
    tdm_load_tile_2d(ldsA[0], aBase, (unsigned)K, (unsigned)M, (unsigned)K);
    tdm_load_tile_2d(ldsB[0], bBase, (unsigned)K, (unsigned)N, (unsigned)K);
  }
#endif

  int buf = 0;
  for (int kt = 0; kt < KT; ++kt) {
#if HAVE_TDM
    if (wave_id == 0) __builtin_amdgcn_s_wait_tensorcnt(0);
    __syncthreads();                           // publish DMA'd tile; retire buf^1
    if (wave_id == 0 && (kt + 1 < KT)) {       // prefetch next K-slab via TDM
      const unsigned long long ko = (unsigned long long)(kt + 1) * 64ull; // 32*2B
      tdm_load_tile_2d(ldsA[buf ^ 1], aBase + ko, (unsigned)K, (unsigned)M, (unsigned)K);
      tdm_load_tile_2d(ldsB[buf ^ 1], bBase + ko, (unsigned)K, (unsigned)N, (unsigned)K);
    }
#else
    {                                          // fallback: vectorized LDS fill
      const int kBase = kt << 5;
#pragma unroll
      for (int i = 0; i < 4; ++i) {
        const int idx = tid + (i << 8);        // 0..1023 ushort4 slots
        const int row = idx >> 3;
        const int kq  = (idx & 7) << 2;
        *(ushort4*)&Asl[buf][row * 48 + kq] =
            *(const ushort4*)(Ag + (size_t)(mBase + row) * K + kBase + kq);
        *(ushort4*)&Bsl[buf][row * 48 + kq] =
            *(const ushort4*)(Btg + (size_t)(nBase + row) * K + kBase + kq);
      }
      __syncthreads();
    }
#endif

    // fragments: contiguous k order, same for A and B -> lanes 0-15 K 0..15, lanes 16-31 K 16..31
    const unsigned short* Ab = Asl[buf];
    const unsigned short* Bb = Bsl[buf];
    const int khalf = (lane >> 4) << 4;
    const int lmod  = lane & 15;
    v16bf a[4], b[2];
#pragma unroll
    for (int fm = 0; fm < 4; ++fm)
      a[fm] = *(const v16bf*)&Ab[(wm * 64 + fm * 16 + lmod) * 48 + khalf];
#pragma unroll
    for (int fn = 0; fn < 2; ++fn)
      b[fn] = *(const v16bf*)&Bb[(wn * 32 + fn * 16 + lmod) * 48 + khalf];

#pragma unroll
    for (int fm = 0; fm < 4; ++fm)
#pragma unroll
      for (int fn = 0; fn < 2; ++fn)
        c[fm][fn] = __builtin_amdgcn_wmma_f32_16x16x32_bf16(
            false, a[fm], false, b[fn], (short)0, c[fm][fn], false, false);

#if !HAVE_TDM
    __syncthreads();
#endif
    buf ^= 1;
  }

  // ------------------------------- epilogue -------------------------------
#pragma unroll
  for (int fm = 0; fm < 4; ++fm) {
#pragma unroll
    for (int fn = 0; fn < 2; ++fn) {
#pragma unroll
      for (int r = 0; r < 8; ++r) {
        const int m = mBase + wm * 64 + fm * 16 + ((lane >> 4) << 3) + r;
        const int n = nBase + wn * 32 + fn * 16 + (lane & 15);
        float cb = c[fm][fn][r] + bias[n];
        if constexpr (EPILOGUE == 1) {
          outp[(size_t)m * N + n] = cb;
        } else {
          // RoPE pairs (n even, n odd) sit in adjacent lanes -> xor-shuffle
          const float pb = __shfl_xor(cb, 1, 32);
          const int l   = m & (SEQ_L - 1);
          const int bb  = m >> 13;               // M = b*L + l
          const int sec = n >> 10;               // 0=q 1=k 2=v
          const int wth = n & 1023;
          const int h   = wth >> 6;
          const int d   = wth & 63;
          const size_t dst = (((size_t)bb * NHEADS + h) * SEQ_L + l) * DHEAD + d;
          const float mval = mask[(size_t)bb * SEQ_L + l];
          if (sec == 2) {
            vp[dst] = cb * mval;
          } else {
            const float fc = fcos[(size_t)l * (DHEAD / 2) + (d >> 1)];
            const float fs = fsin[(size_t)l * (DHEAD / 2) + (d >> 1)];
            const float rot = ((d & 1) == 0) ? (cb * fc - pb * fs)
                                             : (pb * fs + cb * fc);
            const float ep = rot > 0.f ? rot + 1.f : __expf(rot);   // elu+1
            if (sec == 0) qp[dst] = ep;
            else          kp[dst] = ep * mval;
          }
        }
      }
    }
  }
}

// ---------------------------------------------------------------------------
// kv[b,h,d,e] = sum_l k'[l,d] v[l,e] ; ksum[b,h,d] = sum_l k'[l,d].
// Split-K over L (8 splits), fp32 atomics into global accumulators.
// ---------------------------------------------------------------------------
__global__ __launch_bounds__(256, 4)
void kv_reduce_kernel(const float* __restrict__ kp, const float* __restrict__ vp,
                      float* __restrict__ kv, float* __restrict__ ksum)
{
  const int bh = blockIdx.x >> 3;
  const int s  = blockIdx.x & 7;
  const int CH = SEQ_L / 8;

  __shared__ float kl[32][64];
  __shared__ float vl[32][64];

  const int td = threadIdx.x >> 4;
  const int te = threadIdx.x & 15;

  float acc[4][4] = {};
  float ks[4]     = {};

  const float* kb = kp + ((size_t)bh * SEQ_L + (size_t)s * CH) * DHEAD;
  const float* vb = vp + ((size_t)bh * SEQ_L + (size_t)s * CH) * DHEAD;

  for (int l0 = 0; l0 < CH; l0 += 32) {
#pragma unroll
    for (int i = 0; i < 2; ++i) {
      const int idx = threadIdx.x + (i << 8);
      const int r   = idx >> 4;
      const int cq  = (idx & 15) << 2;
      *(float4*)&kl[r][cq] = *(const float4*)(kb + (size_t)(l0 + r) * DHEAD + cq);
      *(float4*)&vl[r][cq] = *(const float4*)(vb + (size_t)(l0 + r) * DHEAD + cq);
    }
    __syncthreads();
#pragma unroll 4
    for (int l = 0; l < 32; ++l) {
      const float4 k4 = *(const float4*)&kl[l][td << 2];
      const float4 v4 = *(const float4*)&vl[l][te << 2];
      const float kd[4] = {k4.x, k4.y, k4.z, k4.w};
      const float ve[4] = {v4.x, v4.y, v4.z, v4.w};
#pragma unroll
      for (int a = 0; a < 4; ++a)
#pragma unroll
        for (int e = 0; e < 4; ++e) acc[a][e] += kd[a] * ve[e];
      if (te == 0) {
#pragma unroll
        for (int a = 0; a < 4; ++a) ks[a] += kd[a];
      }
    }
    __syncthreads();
  }

  float* kvb = kv + (size_t)bh * DHEAD * DHEAD;
#pragma unroll
  for (int a = 0; a < 4; ++a)
#pragma unroll
    for (int e = 0; e < 4; ++e)
      atomicAdd(&kvb[(size_t)((td << 2) + a) * DHEAD + (te << 2) + e], acc[a][e]);
  if (te == 0) {
#pragma unroll
    for (int a = 0; a < 4; ++a)
      atomicAdd(&ksum[(size_t)bh * DHEAD + (td << 2) + a], ks[a]);
  }
}

// ---------------------------------------------------------------------------
// context = q' @ kv ; norm = q' . ksum ; attn = context/(norm+eps) as bf16
// in [B, L, E] layout (A matrix of the output projection GEMM).
// ---------------------------------------------------------------------------
__global__ __launch_bounds__(256, 4)
void ctx_kernel(const float* __restrict__ qp, const float* __restrict__ kv,
                const float* __restrict__ ksum, unsigned short* __restrict__ attn)
{
  const int bh = blockIdx.x;
  const int lt = blockIdx.y;
  const int b  = bh >> 4;
  const int h  = bh & 15;

  __shared__ float kvl[64][64];
  __shared__ float ksl[64];
  __shared__ float ql[4][64];

#pragma unroll
  for (int i = 0; i < 16; ++i) {
    const int idx = threadIdx.x + (i << 8);
    kvl[idx >> 6][idx & 63] = kv[(size_t)bh * 4096 + idx];
  }
  if (threadIdx.x < 64) ksl[threadIdx.x] = ksum[(size_t)bh * DHEAD + threadIdx.x];
  __syncthreads();

  const int ro = threadIdx.x >> 6;
  const int e  = threadIdx.x & 63;
  const float* qbase = qp + ((size_t)bh * SEQ_L + (size_t)lt * 64) * DHEAD;

  for (int r0 = 0; r0 < 64; r0 += 4) {
    ql[ro][e] = qbase[(size_t)(r0 + ro) * DHEAD + e];
    __syncthreads();
    float ctx = 0.f, nrm = 0.f;
#pragma unroll 8
    for (int d = 0; d < 64; ++d) {
      const float qv = ql[ro][d];
      ctx += qv * kvl[d][e];
      nrm += qv * ksl[d];
    }
    const float outv = ctx / (nrm + EPS_LA);
    const int l = lt * 64 + r0 + ro;
    attn[((size_t)b * SEQ_L + l) * EMB + (size_t)h * DHEAD + e] =
        f32_to_bf16_bits(outv);
    __syncthreads();
  }
}

// ---------------------------------------------------------------------------
extern "C" void kernel_launch(void* const* d_in, const int* in_sizes, int n_in,
                              void* d_out, int out_size, void* d_ws, size_t ws_size,
                              hipStream_t stream)
{
  const float* x    = (const float*)d_in[0];
  const float* fcos = (const float*)d_in[1];
  const float* fsin = (const float*)d_in[2];
  const float* mask = (const float*)d_in[3];
  const float* Wqkv = (const float*)d_in[4];
  const float* bqkv = (const float*)d_in[5];
  const float* Wout = (const float*)d_in[6];
  const float* bout = (const float*)d_in[7];
  float* out = (float*)d_out;

  const int M = 4 * SEQ_L;                         // 32768
  const size_t QSZ   = (size_t)M * EMB * 4;        // 134 MB (each of q',k',v)
  const size_t ATTNS = (size_t)M * EMB * 2;        // 67 MB bf16
  const size_t XBS   = (size_t)M * EMB * 2;        // 67 MB bf16
  const size_t WQKVT = (size_t)3 * EMB * EMB * 2;  // 6 MB bf16
  const size_t WOUTT = (size_t)EMB * EMB * 2;      // 2 MB bf16
  const size_t KVS   = (size_t)64 * DHEAD * DHEAD * 4;
  const size_t KSS   = (size_t)64 * DHEAD * 4;

  char* w = (char*)d_ws;
  float*          qp    = (float*)(w);
  float*          kp    = (float*)(w + QSZ);
  float*          vp    = (float*)(w + 2 * QSZ);
  unsigned short* attn  = (unsigned short*)(w + 3 * QSZ);
  unsigned short* xb    = (unsigned short*)(w + 3 * QSZ + ATTNS);
  unsigned short* wqkvt = (unsigned short*)(w + 3 * QSZ + ATTNS + XBS);
  unsigned short* woutt = (unsigned short*)(w + 3 * QSZ + ATTNS + XBS + WQKVT);
  float*          kv    = (float*)(w + 3 * QSZ + ATTNS + XBS + WQKVT + WOUTT);
  float*          ksum  = (float*)(w + 3 * QSZ + ATTNS + XBS + WQKVT + WOUTT + KVS);

  // 0) one-time layout prep: x -> bf16 ; W -> bf16 transposed [N][K]
  const long long n4 = (long long)M * EMB / 4;
  cvt_bf16_kernel<<<dim3((unsigned)((n4 + 255) / 256)), 256, 0, stream>>>(x, xb, n4);
  transpose_cvt_kernel<<<dim3(3 * EMB / 32, EMB / 32), 256, 0, stream>>>(
      Wqkv, wqkvt, EMB, 3 * EMB);
  transpose_cvt_kernel<<<dim3(EMB / 32, EMB / 32), 256, 0, stream>>>(
      Wout, woutt, EMB, EMB);
  (void)hipMemsetAsync(kv, 0, KVS + KSS, stream);

  // 1) qkv = x @ Wqkv + b, fused RoPE / elu+1 / mask -> q', k', v
  wmma_gemm_tdm_kernel<0><<<dim3(3 * EMB / 128, M / 128), 256, 0, stream>>>(
      xb, wqkvt, bqkv, fcos, fsin, mask, qp, kp, vp, nullptr, M, 3 * EMB, EMB);

  // 2) kv = k'^T v ; ksum = sum k'
  kv_reduce_kernel<<<dim3(64 * 8), 256, 0, stream>>>(kp, vp, kv, ksum);

  // 3) context/(norm+eps) -> bf16 [B,L,E]
  ctx_kernel<<<dim3(64, SEQ_L / 64), 256, 0, stream>>>(qp, kv, ksum, attn);

  // 4) out = attn @ Wout + bout (fp32)
  wmma_gemm_tdm_kernel<1><<<dim3(EMB / 128, M / 128), 256, 0, stream>>>(
      attn, woutt, bout, nullptr, nullptr, nullptr, nullptr, nullptr, nullptr,
      out, M, EMB, EMB);
}
